// AttentionPooling_31842887533292
// MI455X (gfx1250) — matrix operand order repaired and checked by reading, not compiled, not run on hardware
//
#include <hip/hip_runtime.h>
#include <hip/hip_bf16.h>

typedef __attribute__((ext_vector_type(16))) __bf16 v16bf;
typedef __attribute__((ext_vector_type(8)))  __bf16 v8bf;
typedef __attribute__((ext_vector_type(8)))  float  v8f;

// ---------- helpers: order-preserving float<->uint for atomicMax ----------
__device__ __forceinline__ unsigned f2ord(float f) {
    unsigned b = __float_as_uint(f);
    return (b & 0x80000000u) ? ~b : (b | 0x80000000u);
}
__device__ __forceinline__ float ord2f(unsigned u) {
    return __uint_as_float((u & 0x80000000u) ? (u & 0x7FFFFFFFu) : ~u);
}

// ---------- kernel 0: W1 [K=256][J=256] f32 -> W1^T [J][K] bf16 ----------
__global__ void prep_w1t_kernel(const float* __restrict__ W1, __bf16* __restrict__ w1t) {
    int j = blockIdx.x;
    int k = threadIdx.x;
    w1t[j * 256 + k] = (__bf16)W1[k * 256 + j];
}

// ---------- kernel 1: zero output, init segmax/segsum ----------
__global__ void init_kernel(float* __restrict__ out, unsigned* __restrict__ segmax,
                            float* __restrict__ segsum, int nOut, int nSeg) {
    int i = blockIdx.x * blockDim.x + threadIdx.x;
    if (i < nOut) out[i] = 0.0f;
    if (i < nSeg) { segmax[i] = 0u; segsum[i] = 0.0f; }   // ord(0) < ord(any real score)
}

// ---------- kernel 2: scores = tanh(x@W1 + b1) @ W2  via bf16 WMMA ----------
// Block: 256 threads = 8 waves; tile 64 rows x 256 cols of h.
// W1^T is staged into LDS in 4 K-chunks of 64 (32 KB each, double-buffered)
// with GLOBAL_LOAD_ASYNC_TO_LDS_B128 overlapping the WMMAs of the prior chunk.
// LDS B layout: sB[buf][col*72 + klocal] (__bf16), 144B row stride
// (16B-aligned fragments; 36*L mod 64 distinct for the 16 cols of a half-wave).
__global__ __launch_bounds__(256) void gemm_scores_kernel(
    const float* __restrict__ x, const __bf16* __restrict__ w1t,
    const float* __restrict__ b1, const float* __restrict__ W2,
    float* __restrict__ scores, int N)
{
    __shared__ __bf16 sB[2][256 * 72];   // 2 x 36 KB staged W1^T chunks
    __shared__ __bf16 hbuf[64 * 260];    // tanh(h) tile, padded stride
    __shared__ float  s_b1[256];
    __shared__ float  s_w2[1024];

    const int t = threadIdx.x;
    s_b1[t] = b1[t];
    for (int i = t; i < 1024; i += 256) s_w2[i] = W2[i];

    const int wid   = t >> 5;
    const int lane  = t & 31;
    const int mtile = wid & 3;        // 16-row tile
    const int chalf = wid >> 2;       // 128-col half
    const int r0    = blockIdx.x * 64;
    const int hi16  = lane >> 4;
    const int lcol  = lane & 15;

    int rowInBlk = 16 * mtile + (lane & 15);
    long grow = (long)r0 + rowInBlk;
    if (grow >= N) grow = N - 1;
    const float* xrow = x + grow * 256;
    const int kb = hi16 * 8;

    const unsigned ldsB[2] = { (unsigned)(size_t)(void*)&sB[0][0],
                               (unsigned)(size_t)(void*)&sB[1][0] };

    // stage K-chunk c (k = c*64 .. c*64+63, all 256 cols) into LDS buffer
    auto stage = [&](int c, unsigned ldsBase) {
        const __bf16* gsrc = w1t + c * 64;
        #pragma unroll
        for (int i = 0; i < 8; ++i) {
            const int p   = i * 256 + t;        // 2048 16B pieces per chunk
            const int col = p >> 3;
            const int seg = p & 7;
            unsigned loff = ldsBase + (unsigned)(col * 144 + seg * 16);
            unsigned long long ga =
                (unsigned long long)(size_t)(gsrc + (size_t)col * 256 + seg * 8);
            asm volatile("global_load_async_to_lds_b128 %0, %1, off"
                         :: "v"(loff), "v"(ga) : "memory");
        }
    };

    v8f acc[8];
    #pragma unroll
    for (int j = 0; j < 8; ++j) { v8f z = {}; acc[j] = z; }

    stage(0, ldsB[0]);
    asm volatile("s_wait_asynccnt 0x0" ::: "memory");
    __syncthreads();

    #pragma unroll
    for (int c = 0; c < 4; ++c) {
        if (c < 3) stage(c + 1, ldsB[(c + 1) & 1]);   // overlap with compute below
        const __bf16* bbuf = &sB[c & 1][0];

        #pragma unroll
        for (int kh = 0; kh < 2; ++kh) {              // two K=32 steps per chunk
            const int kg = c * 64 + kh * 32;          // global K base
            // A fragment (16-bit A 16x32 layout): f32 loads + convert
            v8f lo = *reinterpret_cast<const v8f*>(xrow + kg + kb);
            v8f hi = *reinterpret_cast<const v8f*>(xrow + kg + kb + 16);
            v16bf a;
            #pragma unroll
            for (int e = 0; e < 8; ++e) { a[e] = (__bf16)lo[e]; a[8 + e] = (__bf16)hi[e]; }

            const int klocal = kh * 32 + hi16 * 8;    // B: table-order K (8h+s, 16+8h+s)
            #pragma unroll
            for (int j = 0; j < 8; ++j) {
                const int col = 128 * chalf + 16 * j + lcol;
                const __bf16* bp = bbuf + col * 72 + klocal;
                v8bf blo = *reinterpret_cast<const v8bf*>(bp);
                v8bf bhi = *reinterpret_cast<const v8bf*>(bp + 16);
                v16bf bfrag = __builtin_shufflevector(blo, bhi,
                    0,1,2,3,4,5,6,7,8,9,10,11,12,13,14,15);
                acc[j] = __builtin_amdgcn_wmma_f32_16x16x32_bf16(
                             false, a, false, bfrag, (short)0, acc[j], false, false);
            }
        }
        asm volatile("s_wait_asynccnt 0x0" ::: "memory");  // next chunk landed
        __syncthreads();                                    // visible to all waves
    }

    // + b1, tanh, stash to LDS (C layout: M = r + 8*hi16, N = lane&15)
    #pragma unroll
    for (int j = 0; j < 8; ++j) {
        #pragma unroll
        for (int r = 0; r < 8; ++r) {
            const int row = 16 * mtile + r + 8 * hi16;
            const int col = 128 * chalf + 16 * j + lcol;
            const float v = acc[j][r] + s_b1[col];
            hbuf[row * 260 + col] = (__bf16)tanhf(v);
        }
    }
    __syncthreads();

    // scores: 64 rows x 4 heads = 256 dot products of length 256
    {
        const int row  = t >> 2;
        const int head = t & 3;
        const __bf16* hr = hbuf + row * 260;
        float s = 0.0f;
        #pragma unroll 8
        for (int k = 0; k < 256; ++k) s += (float)hr[k] * s_w2[k * 4 + head];
        const int gr = r0 + row;
        if (gr < N) scores[(size_t)gr * 4 + head] = s;
    }
}

// ---------- kernel 3: segment max (atomicMax on ordered uint) ----------
__global__ void segmax_kernel(const float* __restrict__ scores, const int* __restrict__ batch,
                              unsigned* __restrict__ segmax, int n4) {
    int i = blockIdx.x * blockDim.x + threadIdx.x;
    if (i < n4) {
        int n = i >> 2, h = i & 3;
        int g = batch[n];
        atomicMax(&segmax[g * 4 + h], f2ord(scores[i]));
    }
}

// ---------- kernel 4: e = exp(s - max), segment sum ----------
__global__ void segexp_kernel(float* __restrict__ scores, const int* __restrict__ batch,
                              const unsigned* __restrict__ segmax, float* __restrict__ segsum,
                              int n4) {
    int i = blockIdx.x * blockDim.x + threadIdx.x;
    if (i < n4) {
        int n = i >> 2, h = i & 3;
        int g = batch[n];
        float m  = ord2f(segmax[g * 4 + h]);
        float ev = __expf(scores[i] - m);
        scores[i] = ev;                      // scores buffer now holds e
        atomicAdd(&segsum[g * 4 + h], ev);
    }
}

// ---------- kernel 5: out[g,:] += coef[n] * x[n,:], coef = 0.25*sum_h e/segsum ----------
__global__ __launch_bounds__(256) void pool_kernel(
    const float* __restrict__ x, const int* __restrict__ batch,
    const float* __restrict__ e, const float* __restrict__ segsum,
    float* __restrict__ out, int N)
{
    __shared__ float s_coef[256];
    __shared__ int   s_g[256];
    const int t  = threadIdx.x;
    const int n0 = blockIdx.x * 256;
    const int n  = n0 + t;
    if (n < N) {
        const int g = batch[n];
        float c = 0.0f;
        #pragma unroll
        for (int h = 0; h < 4; ++h) c += e[(size_t)n * 4 + h] / segsum[g * 4 + h];
        s_coef[t] = 0.25f * c;
        s_g[t]    = g;
    } else {
        s_coef[t] = 0.0f;
        s_g[t]    = -1;
    }
    __syncthreads();

    int cnt = N - n0; if (cnt > 256) cnt = 256;
    float acc = 0.0f;
    int cg = s_g[0];                         // batch is sorted: few flushes per chunk
    for (int i = 0; i < cnt; ++i) {
        if (i + 8 < cnt)                     // look-ahead prefetch of x rows
            __builtin_prefetch(&x[(size_t)(n0 + i + 8) * 256 + t], 0, 1);
        const int g = s_g[i];
        if (g != cg) {
            if (cg >= 0) atomicAdd(&out[(size_t)cg * 256 + t], acc);
            acc = 0.0f; cg = g;
        }
        acc += s_coef[i] * x[(size_t)(n0 + i) * 256 + t];
    }
    if (cg >= 0) atomicAdd(&out[(size_t)cg * 256 + t], acc);
}

extern "C" void kernel_launch(void* const* d_in, const int* in_sizes, int n_in,
                              void* d_out, int out_size, void* d_ws, size_t ws_size,
                              hipStream_t stream) {
    (void)n_in; (void)ws_size;
    const float* x     = (const float*)d_in[0];
    const int*   batch = (const int*)d_in[1];
    const float* W1    = (const float*)d_in[2];
    const float* b1    = (const float*)d_in[3];
    const float* W2    = (const float*)d_in[4];

    const int N = in_sizes[0] / 256;     // 200000
    const int G = out_size / 256;        // 512
    float* out = (float*)d_out;

    // workspace layout
    char* ws = (char*)d_ws;
    __bf16* w1t    = (__bf16*)ws;                                   // 256*256*2 = 128 KB
    float*  scores = (float*)(ws + 131072);                         // N*4 f32 (reused as e)
    size_t  o_smx  = (131072 + (size_t)N * 16 + 255) & ~(size_t)255;
    unsigned* segmax = (unsigned*)(ws + o_smx);                     // G*4
    float*    segsum = (float*)(ws + o_smx + (size_t)G * 16);       // G*4

    const int n4 = N * 4;

    prep_w1t_kernel<<<256, 256, 0, stream>>>(W1, w1t);
    init_kernel<<<(G * 256 + 255) / 256, 256, 0, stream>>>(out, segmax, segsum, G * 256, G * 4);
    gemm_scores_kernel<<<(N + 63) / 64, 256, 0, stream>>>(x, w1t, b1, W2, scores, N);
    segmax_kernel<<<(n4 + 255) / 256, 256, 0, stream>>>(scores, batch, segmax, n4);
    segexp_kernel<<<(n4 + 255) / 256, 256, 0, stream>>>(scores, batch, segmax, segsum, n4);
    pool_kernel<<<(N + 255) / 256, 256, 0, stream>>>(x, batch, scores, segsum, out, N);
}
